// EncoderNTM_28827820491519
// MI455X (gfx1250) — matrix operand order repaired and checked by reading, not compiled, not run on hardware
//
#include <hip/hip_runtime.h>
#include <hip/hip_bf16.h>

// ---------------- problem constants ----------------
#define TSTEPS 256
#define BSZ    256
#define IDIM_C 256
#define CDIM_C 512
#define NMEM   128
#define MMEM   64
#define GDIM   2048          // 4*CDIM
#define KTOT   832           // IDIM + M + CDIM
#define HROWS  384           // 268 head rows padded to 384 (3*128)
#define OSW    576           // CDIM + M

// finals layout offsets (floats) inside fin = d_out + T*B*OSW
#define FH_OFF  0
#define FC_OFF  (BSZ*CDIM_C)
#define FR_OFF  (2*BSZ*CDIM_C)
#define FWR_OFF (FR_OFF + BSZ*MMEM)
#define FWW_OFF (FWR_OFF + BSZ*NMEM)
#define FM_OFF  (FWW_OFF + BSZ*NMEM)

typedef __attribute__((ext_vector_type(16))) __bf16 v16bf;
typedef __attribute__((ext_vector_type(8)))  float  v8f;
typedef unsigned int u32x4 __attribute__((ext_vector_type(4)));
typedef int          i32x8 __attribute__((ext_vector_type(8)));
typedef int          i32x4 __attribute__((ext_vector_type(4)));

union Frag16 { v16bf v; unsigned short u[16]; u32x4 q[2]; };

__device__ inline unsigned short f2bf(float f) {
    unsigned int u = __float_as_uint(f);
    unsigned int r = u + 0x7FFFu + ((u >> 16) & 1u);
    return (unsigned short)(r >> 16);
}
__device__ inline float sigmoidf_(float x) { return 1.f / (1.f + __expf(-x)); }
__device__ inline float softplusf_(float x) { return (x > 20.f) ? x : log1pf(__expf(x)); }

// ---------------- prep kernels ----------------
// Swizzled weight layout: WgSw[((kb*GDIM)+j)*32 + kk] = W[j][kb*32+kk]  (bf16)
// -> a WMMA B-fragment (col j, lane-half h) is 2 contiguous 16B chunks in global.
__global__ void prep_wgt(const float* __restrict__ W_ih, const float* __restrict__ W_hh,
                         unsigned short* __restrict__ WgSw) {
    int idx = blockIdx.x * 256 + threadIdx.x;          // over KTOT*GDIM
    if (idx >= KTOT * GDIM) return;
    int kb  = idx / (GDIM * 32);
    int rem = idx - kb * (GDIM * 32);
    int j   = rem >> 5;
    int kk  = rem & 31;
    int k   = kb * 32 + kk;
    float v = (k < IDIM_C + MMEM) ? W_ih[j * (IDIM_C + MMEM) + k]
                                  : W_hh[j * CDIM_C + (k - (IDIM_C + MMEM))];
    WgSw[idx] = f2bf(v);
}

__global__ void prep_whead(const float* __restrict__ Wr, const float* __restrict__ Ww,
                           unsigned short* __restrict__ WhSw) {
    int idx = blockIdx.x * 256 + threadIdx.x;          // over CDIM*HROWS
    if (idx >= CDIM_C * HROWS) return;
    int kb  = idx / (HROWS * 32);
    int rem = idx - kb * (HROWS * 32);
    int j   = rem >> 5;
    int kk  = rem & 31;
    int k   = kb * 32 + kk;
    float v = 0.f;
    if (j < 70)       v = Wr[j * CDIM_C + k];
    else if (j < 268) v = Ww[(j - 70) * CDIM_C + k];
    WhSw[idx] = f2bf(v);
}

__global__ void prep_bias(const float* __restrict__ br, const float* __restrict__ bw,
                          float* __restrict__ bhead) {
    int idx = blockIdx.x * 256 + threadIdx.x;
    if (idx >= HROWS) return;
    float v = 0.f;
    if (idx < 70)       v = br[idx];
    else if (idx < 268) v = bw[idx - 70];
    bhead[idx] = v;
}

__global__ void prep_state(const float* __restrict__ h0, const float* __restrict__ c0,
                           const float* __restrict__ r0, const float* __restrict__ mem_bias,
                           float* __restrict__ h_st, unsigned short* __restrict__ h_bf,
                           float* __restrict__ c_st, float* __restrict__ r_st,
                           unsigned short* __restrict__ r_bf,
                           float* __restrict__ w_r, float* __restrict__ w_w,
                           float* __restrict__ mem) {
    int idx = blockIdx.x * 256 + threadIdx.x;          // grid covers BSZ*NMEM*MMEM
    if (idx < BSZ * CDIM_C) {
        int jc = idx & (CDIM_C - 1);
        float h = h0[jc];
        h_st[idx] = h; h_bf[idx] = f2bf(h); c_st[idx] = c0[jc];
    }
    if (idx < BSZ * MMEM) {
        float r = r0[idx & (MMEM - 1)];
        r_st[idx] = r; r_bf[idx] = f2bf(r);
    }
    if (idx < BSZ * NMEM) { w_r[idx] = 0.f; w_w[idx] = 0.f; }
    if (idx < BSZ * NMEM * MMEM) mem[idx] = mem_bias[idx & (NMEM * MMEM - 1)];
}

// ---------------- WMMA fragment helpers ----------------
// A fragment (16x32 bf16): lane(row=lane&15, half=lane>>4) elements are one
// contiguous K-run: [half*16 .. half*16+15].
__device__ inline void load_afrag(Frag16& af, const unsigned short (*sA)[40],
                                  int rowbase, int lane) {
    int row  = rowbase + (lane & 15);
    int half = lane >> 4;
    af.q[0] = *(const u32x4*)&sA[row][half * 16];
    af.q[1] = *(const u32x4*)&sA[row][half * 16 + 8];
}
// B fragment (32x16 bf16) from swizzled weights: col j's 32 K-values are
// contiguous; lane-half picks 16 of them (two 16B chunks).
__device__ inline void load_bfrag(Frag16& bf_, const unsigned short* __restrict__ Wsw,
                                  size_t kbBase, int j, int lane) {
    int half = lane >> 4;
    const unsigned short* src = Wsw + kbBase + (size_t)j * 32 + (half << 4);
    bf_.q[0] = *(const u32x4*)src;
    bf_.q[1] = *(const u32x4*)(src + 8);
}

// ---------------- gates GEMM: [emb|r|h](256x832) x W -> gates(256x2048) ----------------
__global__ __launch_bounds__(256) void gates_gemm(
    const float* __restrict__ embs_t,            // B x IDIM (f32)
    const unsigned short* __restrict__ r_bf,     // B x M
    const unsigned short* __restrict__ h_bf,     // B x CDIM
    const unsigned short* __restrict__ WgSw,     // swizzled (KTOT/32, GDIM, 32) bf16
    const float* __restrict__ b_ih, const float* __restrict__ b_hh,
    float* __restrict__ gates)                   // B x GDIM
{
    __shared__ unsigned short sA[64][40];        // 64 b-rows x 32 k (+pad)
    const int tid  = threadIdx.x;
    const int lane = tid & 31;
    const int wave = tid >> 5;
    const int mw   = wave & 3;                   // m sub-tile (16 rows)
    const int nw   = wave >> 2;                  // n half (64 cols)
    const int j0   = blockIdx.x * 128;
    const int b0   = blockIdx.y * 64;

    const v8f vz = {0.f,0.f,0.f,0.f,0.f,0.f,0.f,0.f};
    v8f acc[4];
    #pragma unroll
    for (int i = 0; i < 4; ++i) acc[i] = vz;

    for (int k0 = 0; k0 < KTOT; k0 += 32) {
        if (k0 + 32 < KTOT)
            __builtin_prefetch(WgSw + ((size_t)((k0 >> 5) + 1) * GDIM + j0) * 32, 0, 1);
        // stage A (64x32): row = tid>>2, 8 contiguous k at (tid&3)*8
        {
            int row = tid >> 2;
            int kk  = (tid & 3) * 8;
            int b   = b0 + row;
            int kg  = k0 + kk;                   // k-tiles align with 256/320 boundaries
            if (kg < IDIM_C) {
                const float* src = embs_t + (size_t)b * IDIM_C + kg;
                #pragma unroll
                for (int e = 0; e < 8; ++e) sA[row][kk + e] = f2bf(src[e]);
            } else if (kg < IDIM_C + MMEM) {
                const unsigned short* src = r_bf + (size_t)b * MMEM + (kg - IDIM_C);
                #pragma unroll
                for (int e = 0; e < 8; ++e) sA[row][kk + e] = src[e];
            } else {
                const unsigned short* src = h_bf + (size_t)b * CDIM_C + (kg - IDIM_C - MMEM);
                #pragma unroll
                for (int e = 0; e < 8; ++e) sA[row][kk + e] = src[e];
            }
        }
        __syncthreads();
        Frag16 af;
        load_afrag(af, sA, mw * 16, lane);
        size_t kbBase = (size_t)(k0 >> 5) * GDIM * 32;
        #pragma unroll
        for (int i = 0; i < 4; ++i) {
            Frag16 bf_;
            load_bfrag(bf_, WgSw, kbBase, j0 + nw * 64 + i * 16 + (lane & 15), lane);
            acc[i] = __builtin_amdgcn_wmma_f32_16x16x32_bf16(
                         false, af.v, false, bf_.v, (short)0, acc[i], false, false);
        }
        __syncthreads();
    }
    // store D (8 VGPR f32 layout: row = v + 8*(lane>>4), col = lane&15)
    {
        int colb = lane & 15;
        int rowb = (lane >> 4) * 8;
        #pragma unroll
        for (int i = 0; i < 4; ++i) {
            int j = j0 + nw * 64 + i * 16 + colb;
            float bias = b_ih[j] + b_hh[j];
            #pragma unroll
            for (int v = 0; v < 8; ++v) {
                int b = b0 + mw * 16 + rowb + v;
                gates[(size_t)b * GDIM + j] = acc[i][v] + bias;
            }
        }
    }
}

// ---------------- head GEMM: h(256x512) x [Wr;Ww] -> o_head(256x384) ----------------
__global__ __launch_bounds__(256) void head_gemm(
    const unsigned short* __restrict__ h_bf,
    const unsigned short* __restrict__ WhSw,     // swizzled (16, HROWS, 32) bf16
    const float* __restrict__ bhead,
    float* __restrict__ o_head)
{
    __shared__ unsigned short sA[64][40];
    const int tid  = threadIdx.x;
    const int lane = tid & 31;
    const int wave = tid >> 5;
    const int mw   = wave & 3;
    const int nw   = wave >> 2;
    const int j0   = blockIdx.x * 128;
    const int b0   = blockIdx.y * 64;

    const v8f vz = {0.f,0.f,0.f,0.f,0.f,0.f,0.f,0.f};
    v8f acc[4];
    #pragma unroll
    for (int i = 0; i < 4; ++i) acc[i] = vz;

    for (int k0 = 0; k0 < CDIM_C; k0 += 32) {
        {
            int row = tid >> 2;
            int kk  = (tid & 3) * 8;
            const unsigned short* src = h_bf + (size_t)(b0 + row) * CDIM_C + k0 + kk;
            #pragma unroll
            for (int e = 0; e < 8; ++e) sA[row][kk + e] = src[e];
        }
        __syncthreads();
        Frag16 af;
        load_afrag(af, sA, mw * 16, lane);
        size_t kbBase = (size_t)(k0 >> 5) * HROWS * 32;
        #pragma unroll
        for (int i = 0; i < 4; ++i) {
            Frag16 bf_;
            load_bfrag(bf_, WhSw, kbBase, j0 + nw * 64 + i * 16 + (lane & 15), lane);
            acc[i] = __builtin_amdgcn_wmma_f32_16x16x32_bf16(
                         false, af.v, false, bf_.v, (short)0, acc[i], false, false);
        }
        __syncthreads();
    }
    {
        int colb = lane & 15;
        int rowb = (lane >> 4) * 8;
        #pragma unroll
        for (int i = 0; i < 4; ++i) {
            int j = j0 + nw * 64 + i * 16 + colb;
            float bias = bhead[j];
            #pragma unroll
            for (int v = 0; v < 8; ++v) {
                int b = b0 + mw * 16 + rowb + v;
                o_head[(size_t)b * HROWS + j] = acc[i][v] + bias;
            }
        }
    }
}

// ---------------- LSTM pointwise ----------------
__global__ __launch_bounds__(256) void lstm_point(
    const float* __restrict__ gates,
    float* __restrict__ c_st, float* __restrict__ h_st,
    unsigned short* __restrict__ h_bf,
    float* __restrict__ os_t)                    // B x OSW, step-offset applied
{
    int idx = blockIdx.x * 256 + threadIdx.x;    // < B*CDIM
    int b  = idx >> 9;
    int jc = idx & (CDIM_C - 1);
    const float* gb = gates + (size_t)b * GDIM;
    float ig = gb[jc];
    float fg = gb[CDIM_C + jc];
    float gg = gb[2 * CDIM_C + jc];
    float og = gb[3 * CDIM_C + jc];
    float c  = sigmoidf_(fg) * c_st[idx] + sigmoidf_(ig) * tanhf(gg);
    float h  = sigmoidf_(og) * tanhf(c);
    c_st[idx] = c;
    h_st[idx] = h;
    h_bf[idx] = f2bf(h);
    os_t[(size_t)b * OSW + jc] = h;
}

// ---------------- NTM addressing (shared by both heads) ----------------
__device__ float do_address(int tid, const float* sMem, const float* sK, float nb,
                            float beta, float g, float s0, float s1, float s2, float gamma,
                            const float* wprev, float* sWg, float* sRed)
{
    float dot = 0.f, nasq = 0.f;
    const float* arow = sMem + tid * MMEM;
    #pragma unroll 8
    for (int m = 0; m < MMEM; ++m) {
        float a = arow[m] + 1e-16f;
        dot += a * sK[m];
        nasq += a * a;
    }
    float na = fmaxf(sqrtf(nasq), 1e-8f);
    float score = beta * dot / (na * nb);
    sRed[tid] = score; __syncthreads();
    for (int off = 64; off > 0; off >>= 1) {
        if (tid < off) sRed[tid] = fmaxf(sRed[tid], sRed[tid + off]);
        __syncthreads();
    }
    float mx = sRed[0]; __syncthreads();
    float ev = __expf(score - mx);
    sRed[tid] = ev; __syncthreads();
    for (int off = 64; off > 0; off >>= 1) {
        if (tid < off) sRed[tid] += sRed[tid + off];
        __syncthreads();
    }
    float wc = ev / sRed[0]; __syncthreads();
    float wg = g * wc + (1.f - g) * wprev[tid];
    sWg[tid] = wg; __syncthreads();
    float sh = s0 * sWg[(tid + NMEM - 1) & (NMEM - 1)] + s1 * sWg[tid]
             + s2 * sWg[(tid + 1) & (NMEM - 1)];
    float w = powf(fmaxf(sh, 0.f), gamma);
    sRed[tid] = w; __syncthreads();
    for (int off = 64; off > 0; off >>= 1) {
        if (tid < off) sRed[tid] += sRed[tid + off];
        __syncthreads();
    }
    float wn = w / (sRed[0] + 1e-16f);
    __syncthreads();
    return wn;
}

__global__ __launch_bounds__(128) void address_step(
    const float* __restrict__ o_head,            // B x HROWS
    float* __restrict__ mem,                     // B x N x M
    float* __restrict__ w_r, float* __restrict__ w_w,
    float* __restrict__ r_st, unsigned short* __restrict__ r_bf,
    const float* __restrict__ h_st, const float* __restrict__ c_st,
    const int* __restrict__ lens, int t,
    float* __restrict__ os_t,                    // B x OSW (step-offset applied)
    float* __restrict__ fin)                     // finals base
{
    __shared__ float sMem[NMEM * MMEM];
    __shared__ float sK[MMEM];
    __shared__ float sE[MMEM];
    __shared__ float sAdd[MMEM];
    __shared__ float sRed[NMEM];
    __shared__ float sWg[NMEM];
    const int b = blockIdx.x;
    const int tid = threadIdx.x;
    float* memb = mem + (size_t)b * NMEM * MMEM;

    // ---- stage this batch element's 32KB memory matrix into LDS ----
#if defined(__gfx1250__) && __has_builtin(__builtin_amdgcn_tensor_load_to_lds)
    // Tensor Data Mover: 1-D tile of 8192 f32, ISA 8.3/8.4 descriptor.
    if (tid < 32) {                              // wave 0 issues the TDM op
        unsigned lds_addr = (unsigned)(size_t)(void*)sMem;     // LDS aperture: offset in addr[31:0]
        unsigned long long ga = (unsigned long long)(size_t)memb;
        u32x4 g0;
        g0.x = 1u;                                             // count=1 (valid D#)
        g0.y = lds_addr;                                       // lds_addr
        g0.z = (unsigned)(ga & 0xFFFFFFFFu);                   // global_addr[31:0]
        g0.w = (unsigned)((ga >> 32) & 0x01FFFFFFu)            // global_addr[56:32]
             | (2u << 30);                                     // type=2 ("image")
        i32x8 g1;
        g1[0] = (int)(2u << 16);                               // data_size=2 -> 4 bytes
        g1[1] = (int)((8192u & 0xFFFFu) << 16);                // tensor_dim0[15:0]
        g1[2] = (int)((8192u >> 16) | (1u << 16));             // tensor_dim0[31:16] | tensor_dim1[15:0]=1
        g1[3] = (int)(8192u << 16);                            // tile_dim0=8192
        g1[4] = 1;                                             // tile_dim1=1, tile_dim2=0
        g1[5] = 8192;                                          // tensor_dim0_stride[31:0]
        g1[6] = 0;
        g1[7] = 0;
        i32x4 z4 = {0, 0, 0, 0};
        i32x8 z8 = {0, 0, 0, 0, 0, 0, 0, 0};
        __builtin_amdgcn_tensor_load_to_lds(g0, g1, z4, z4, z8, 0);
        __builtin_amdgcn_s_wait_tensorcnt(0);
    }
#else
    for (int i = tid; i < NMEM * MMEM; i += 128) sMem[i] = memb[i];
#endif
    __syncthreads();

    const float* oh = o_head + (size_t)b * HROWS;

    // ---- read head ----
    if (tid < MMEM) sK[tid] = oh[tid] + 1e-16f;
    __syncthreads();
    float beta  = softplusf_(oh[64]);
    float g     = sigmoidf_(oh[65]);
    float a0 = oh[66], a1 = oh[67], a2 = oh[68];
    float m3 = fmaxf(fmaxf(a0, a1), a2);
    float x0 = __expf(a0 - m3), x1 = __expf(a1 - m3), x2 = __expf(a2 - m3);
    float si = 1.f / (x0 + x1 + x2);
    float gamma = 1.f + softplusf_(oh[69]);
    float nbsq = 0.f;
    #pragma unroll 8
    for (int m = 0; m < MMEM; ++m) nbsq += sK[m] * sK[m];
    float nb = fmaxf(sqrtf(nbsq), 1e-8f);
    float wr_new = do_address(tid, sMem, sK, nb, beta, g, x0 * si, x1 * si, x2 * si, gamma,
                              w_r + (size_t)b * NMEM, sWg, sRed);
    w_r[(size_t)b * NMEM + tid] = wr_new;
    sWg[tid] = wr_new; __syncthreads();

    float rm = 0.f;
    if (tid < MMEM) {
        #pragma unroll 8
        for (int n = 0; n < NMEM; ++n) rm += sWg[n] * sMem[n * MMEM + tid];
        r_st[(size_t)b * MMEM + tid] = rm;
        r_bf[(size_t)b * MMEM + tid] = f2bf(rm);
        os_t[(size_t)b * OSW + CDIM_C + tid] = rm;
    }
    __syncthreads();

    // ---- write head ----
    if (tid < MMEM) {
        sK[tid]   = oh[70 + tid] + 1e-16f;
        sE[tid]   = sigmoidf_(oh[140 + tid]);
        sAdd[tid] = oh[204 + tid];
    }
    __syncthreads();
    float betaw = softplusf_(oh[134]);
    float gw    = sigmoidf_(oh[135]);
    float b0s = oh[136], b1s = oh[137], b2s = oh[138];
    float mw3 = fmaxf(fmaxf(b0s, b1s), b2s);
    float y0 = __expf(b0s - mw3), y1 = __expf(b1s - mw3), y2 = __expf(b2s - mw3);
    float siw = 1.f / (y0 + y1 + y2);
    float gammaw = 1.f + softplusf_(oh[139]);
    float nbwsq = 0.f;
    #pragma unroll 8
    for (int m = 0; m < MMEM; ++m) nbwsq += sK[m] * sK[m];
    float nbw = fmaxf(sqrtf(nbwsq), 1e-8f);
    float ww_new = do_address(tid, sMem, sK, nbw, betaw, gw, y0 * siw, y1 * siw, y2 * siw, gammaw,
                              w_w + (size_t)b * NMEM, sWg, sRed);
    w_w[(size_t)b * NMEM + tid] = ww_new;
    sWg[tid] = ww_new; __syncthreads();

    // memory update: mem = mem*(1 - w_w e^T) + w_w a^T
    for (int i = tid; i < NMEM * MMEM; i += 128) {
        int n = i >> 6, m = i & (MMEM - 1);
        float v = sMem[i] * (1.f - sWg[n] * sE[m]) + sWg[n] * sAdd[m];
        sMem[i] = v;
        memb[i] = v;
    }
    __syncthreads();

    // ---- finals capture (block-uniform predicate) ----
    if (t == lens[b] - 1) {
        for (int i = tid; i < CDIM_C; i += 128) {
            fin[FH_OFF + (size_t)b * CDIM_C + i] = h_st[(size_t)b * CDIM_C + i];
            fin[FC_OFF + (size_t)b * CDIM_C + i] = c_st[(size_t)b * CDIM_C + i];
        }
        if (tid < MMEM) fin[FR_OFF + (size_t)b * MMEM + tid] = rm;
        fin[FWR_OFF + (size_t)b * NMEM + tid] = wr_new;
        fin[FWW_OFF + (size_t)b * NMEM + tid] = ww_new;
        for (int i = tid; i < NMEM * MMEM; i += 128)
            fin[FM_OFF + (size_t)b * NMEM * MMEM + i] = sMem[i];
    }
}

// ---------------- launcher ----------------
extern "C" void kernel_launch(void* const* d_in, const int* in_sizes, int n_in,
                              void* d_out, int out_size, void* d_ws, size_t ws_size,
                              hipStream_t stream) {
    const float* embs     = (const float*)d_in[0];
    const float* mem_bias = (const float*)d_in[1];
    const float* W_ih     = (const float*)d_in[2];
    const float* W_hh     = (const float*)d_in[3];
    const float* b_ih     = (const float*)d_in[4];
    const float* b_hh     = (const float*)d_in[5];
    const float* Wr       = (const float*)d_in[6];
    const float* br       = (const float*)d_in[7];
    const float* Ww       = (const float*)d_in[8];
    const float* bw       = (const float*)d_in[9];
    const float* h0       = (const float*)d_in[10];
    const float* c0       = (const float*)d_in[11];
    const float* r0       = (const float*)d_in[12];
    const int*   lens     = (const int*)d_in[13];

    char* ws = (char*)d_ws;
    size_t off = 0;
    auto carve = [&](size_t bytes) -> void* {
        void* p = ws + off;
        off = (off + bytes + 255) & ~(size_t)255;
        return p;
    };
    unsigned short* WgSw    = (unsigned short*)carve((size_t)KTOT * GDIM * 2);
    unsigned short* WhSw    = (unsigned short*)carve((size_t)CDIM_C * HROWS * 2);
    float*          bhead   = (float*)carve(HROWS * 4);
    float*          h_st    = (float*)carve((size_t)BSZ * CDIM_C * 4);
    float*          c_st    = (float*)carve((size_t)BSZ * CDIM_C * 4);
    unsigned short* h_bf    = (unsigned short*)carve((size_t)BSZ * CDIM_C * 2);
    float*          r_st    = (float*)carve((size_t)BSZ * MMEM * 4);
    unsigned short* r_bf    = (unsigned short*)carve((size_t)BSZ * MMEM * 2);
    float*          w_r     = (float*)carve((size_t)BSZ * NMEM * 4);
    float*          w_w     = (float*)carve((size_t)BSZ * NMEM * 4);
    float*          mem     = (float*)carve((size_t)BSZ * NMEM * MMEM * 4);
    float*          gates   = (float*)carve((size_t)BSZ * GDIM * 4);
    float*          o_head  = (float*)carve((size_t)BSZ * HROWS * 4);

    float* out = (float*)d_out;
    float* fin = out + (size_t)TSTEPS * BSZ * OSW;

    prep_wgt<<<(KTOT * GDIM + 255) / 256, 256, 0, stream>>>(W_ih, W_hh, WgSw);
    prep_whead<<<(CDIM_C * HROWS + 255) / 256, 256, 0, stream>>>(Wr, Ww, WhSw);
    prep_bias<<<2, 256, 0, stream>>>(br, bw, bhead);
    prep_state<<<(BSZ * NMEM * MMEM + 255) / 256, 256, 0, stream>>>(
        h0, c0, r0, mem_bias, h_st, h_bf, c_st, r_st, r_bf, w_r, w_w, mem);

    for (int t = 0; t < TSTEPS; ++t) {
        const float* embs_t = embs + (size_t)t * BSZ * IDIM_C;
        float* os_t = out + (size_t)t * BSZ * OSW;
        gates_gemm<<<dim3(GDIM / 128, BSZ / 64), 256, 0, stream>>>(
            embs_t, r_bf, h_bf, WgSw, b_ih, b_hh, gates);
        lstm_point<<<(BSZ * CDIM_C) / 256, 256, 0, stream>>>(gates, c_st, h_st, h_bf, os_t);
        head_gemm<<<dim3(HROWS / 128, BSZ / 64), 256, 0, stream>>>(h_bf, WhSw, bhead, o_head);
        address_step<<<BSZ, 128, 0, stream>>>(o_head, mem, w_r, w_w, r_st, r_bf,
                                              h_st, c_st, lens, t, os_t, fin);
    }
}
